// ConditionalRealNVP_31722628448866
// MI455X (gfx1250) — matrix operand order repaired and checked by reading, not compiled, not run on hardware
//
#include <hip/hip_runtime.h>

// ---------------------------------------------------------------------------
// ConditionalRealNVP fused implementation for gfx1250 (MI455X, wave32, WMMA)
//
// Compute-bound (~739 GFLOP vs ~0.3 GB min HBM traffic): all GEMMs run on
// v_wmma_f32_16x16x32_bf16. All operands are pre-converted to bf16 once per
// call (weights additionally transposed to N-major) so the GEMM inner loop is
// pure {async copy -> ds_load_b128 fragments -> WMMA}. 128x128 block tiles,
// 8 waves, 8 WMMAs/wave/K-step, double-buffered LDS with ASYNCcnt waits.
// ---------------------------------------------------------------------------
typedef __attribute__((ext_vector_type(16))) __bf16 v16bf;
typedef __attribute__((ext_vector_type(8)))  float  v8f;
typedef __attribute__((ext_vector_type(4)))  int    v4i;
typedef unsigned short ushort_t;

#define B_   8192
#define D_   256
#define H_   1024
#define A_   128
#define BD_  128
#define IN_  3200
#define L_   8

#define TILE_M 128
#define TILE_N 128
#define KSTEP  32
#define APITCH 40   // ushort pitch: 80 bytes -> 16B-aligned rows, bank-skewed

#if defined(__has_builtin)
#  if __has_builtin(__builtin_amdgcn_global_load_async_to_lds_b128) && \
      __has_builtin(__builtin_amdgcn_s_wait_asynccnt)
#    define USE_ASYNC 1
#  endif
#endif

__device__ __forceinline__ unsigned short f2bf(float f) {
  unsigned u = __float_as_uint(f);
  u += 0x7FFFu + ((u >> 16) & 1u);          // round-to-nearest-even truncate
  return (unsigned short)(u >> 16);
}

#ifdef USE_ASYNC
// 16B per-lane async copy global->LDS (tracked by ASYNCcnt)
__device__ __forceinline__ void async_cp16(const void* g, void* l) {
  __attribute__((address_space(1))) v4i* gp =
      (__attribute__((address_space(1))) v4i*)(unsigned long long)g;
  __attribute__((address_space(3))) v4i* lp =
      (__attribute__((address_space(3))) v4i*)(unsigned)(unsigned long long)l;
  __builtin_amdgcn_global_load_async_to_lds_b128(gp, lp, 0, 0);
}
#endif

union Frag { uint4 u[2]; v16bf v; };

// ---------------------------------------------------------------------------
// Generic fused GEMM:  out = act( A_bf16 @ Wb_bf16^T + bias )
//   Wb: bf16, N-major (pitch K) — pre-transposed/converted
//   a_mode 0: A = Abf, bf16, pitch K
//   a_mode 1: A = virtual concat [xab (bf16, pitch 128) | Eb (bf16, pitch 3072)]
//   act 0: silu -> bf16 out;  act 1: clip(+-3) -> f32;  act 2: none -> f32
// ---------------------------------------------------------------------------
__global__ __launch_bounds__(256)
void fused_gemm_kernel(const ushort_t* __restrict__ Abf,
                       const ushort_t* __restrict__ xab,
                       const ushort_t* __restrict__ Eb,
                       const ushort_t* __restrict__ Wb,
                       const float* __restrict__ bias,
                       void* __restrict__ out,
                       int N, int K, int act, int a_mode) {
  __shared__ __align__(16) ushort_t aT[2][TILE_M][APITCH];  // [m][k]
  __shared__ __align__(16) ushort_t wT[2][TILE_N][APITCH];  // [n][k]

  const int tid  = threadIdx.x;
  const int lane = tid & 31;
  const int wave = tid >> 5;
  const int wm   = wave & 3;       // 4 waves along M (32 rows each)
  const int wn   = wave >> 2;      // 2 waves along N (64 cols each)
  const bool lo  = lane < 16;
  const int l15  = lane & 15;

  const int m0 = blockIdx.x * TILE_M;
  const int n0 = blockIdx.y * TILE_N;

  v8f acc[2][4] = {};

  // stage one 128x32 bf16 tile: 512 chunks of 8 elems, 2 chunks/thread
  auto stageA = [&](int buf, int kk) {
#pragma unroll
    for (int i = 0; i < 2; ++i) {
      const int lin = tid * 2 + i;
      const int r   = lin >> 2;
      const int c8  = (lin & 3) * 8;
      const ushort_t* g;
      if (a_mode == 0) {
        g = Abf + (size_t)(m0 + r) * K + kk + c8;
      } else {
        const int gc = kk + c8;           // chunk never straddles the 128 split
        g = (gc < 128) ? xab + (size_t)(m0 + r) * 128 + gc
                       : Eb  + (size_t)(m0 + r) * 3072 + (gc - 128);
      }
#ifdef USE_ASYNC
      async_cp16(g, &aT[buf][r][c8]);
#else
      *(uint4*)&aT[buf][r][c8] = *(const uint4*)g;
#endif
    }
  };
  auto stageW = [&](int buf, int kk) {
#pragma unroll
    for (int i = 0; i < 2; ++i) {
      const int lin = tid * 2 + i;
      const int n   = lin >> 2;
      const int c8  = (lin & 3) * 8;
      const ushort_t* g = Wb + (size_t)(n0 + n) * K + kk + c8;
#ifdef USE_ASYNC
      async_cp16(g, &wT[buf][n][c8]);
#else
      *(uint4*)&wT[buf][n][c8] = *(const uint4*)g;
#endif
    }
  };

  stageA(0, 0);
  stageW(0, 0);

  int ib = 0;
  for (int kk = 0; kk < K; kk += KSTEP) {
    const bool pf = (kk + KSTEP) < K;
    if (pf) { stageA(ib ^ 1, kk + KSTEP); stageW(ib ^ 1, kk + KSTEP); }
#ifdef USE_ASYNC
    // 4 ops/thread per stage; leaving the newest stage in flight guarantees
    // the current buffer's (older, in-order) copies have completed.
    if (pf) __builtin_amdgcn_s_wait_asynccnt(4);
    else    __builtin_amdgcn_s_wait_asynccnt(0);
#endif
    __syncthreads();

    // ---- WMMA fragments per the CDNA5 16-bit A/B VGPR layout --------------
    // A 16x32: lanes 0-15 hold K {0..7,16..23}, lanes 16-31 hold {8..15,24..31}
    Frag af[2];
#pragma unroll
    for (int mf = 0; mf < 2; ++mf) {
      const uint4* arow = (const uint4*)&aT[ib][wm * 32 + mf * 16 + l15][0];
      af[mf].u[0] = arow[lo ? 0 : 1];
      af[mf].u[1] = arow[lo ? 2 : 3];
    }
    // B 32x16: lanes 0-15 hold K {0..7,16..23}, lanes 16-31 hold {8..15,24..31} (N = lane&15)
    Frag bfm[4];
#pragma unroll
    for (int ns = 0; ns < 4; ++ns) {
      const uint4* brow = (const uint4*)&wT[ib][wn * 64 + ns * 16 + l15][0];
      bfm[ns].u[0] = brow[lo ? 0 : 1];
      bfm[ns].u[1] = brow[lo ? 2 : 3];
    }
#pragma unroll
    for (int mf = 0; mf < 2; ++mf)
#pragma unroll
      for (int ns = 0; ns < 4; ++ns)
        acc[mf][ns] = __builtin_amdgcn_wmma_f32_16x16x32_bf16(
            false, af[mf].v, false, bfm[ns].v, (short)0, acc[mf][ns],
            false, false);
    __syncthreads();
    ib ^= 1;
  }

  // ---- epilogue: bias + activation; C/D layout M=(r | r+8), N=lane&15 -----
#pragma unroll
  for (int mf = 0; mf < 2; ++mf) {
#pragma unroll
    for (int ns = 0; ns < 4; ++ns) {
      const int gn  = n0 + wn * 64 + ns * 16 + l15;
      const float b = bias[gn];
#pragma unroll
      for (int r = 0; r < 8; ++r) {
        const int gm = m0 + wm * 32 + mf * 16 + (lo ? r : r + 8);
        float v = acc[mf][ns][r] + b;
        if (act == 0) {
          v = v * (1.0f / (1.0f + __expf(-v)));                // silu
          ((ushort_t*)out)[(size_t)gm * N + gn] = f2bf(v);
        } else {
          if (act == 1) v = fminf(3.0f, fmaxf(-3.0f, v));      // clip
          ((float*)out)[(size_t)gm * N + gn] = v;
        }
      }
    }
  }
}

// Tiled transpose + fp32->bf16: in (K x N) -> out (N x K), per layer (z-dim)
__global__ __launch_bounds__(256)
void wconv_kernel(const float* __restrict__ in, ushort_t* __restrict__ out,
                  int K, int N) {
  __shared__ float tile[32][33];
  const size_t base = (size_t)blockIdx.z * K * N;
  const int kb = blockIdx.y * 32, nb = blockIdx.x * 32;
  const int tx = threadIdx.x & 31, ty = threadIdx.x >> 5;   // 32 x 8
#pragma unroll
  for (int i = 0; i < 32; i += 8)
    tile[ty + i][tx] = in[base + (size_t)(kb + ty + i) * N + nb + tx];
  __syncthreads();
#pragma unroll
  for (int i = 0; i < 32; i += 8)
    out[base + (size_t)(nb + ty + i) * K + kb + tx] = f2bf(tile[tx][ty + i]);
}

// Concat [ctx | t_emb | cond_emb] -> bf16 E (B x 3072)
__global__ __launch_bounds__(256)
void econv_kernel(const float* __restrict__ ctx, const float* __restrict__ te,
                  const float* __restrict__ ce, ushort_t* __restrict__ E) {
  const size_t t = (size_t)blockIdx.x * 256 + threadIdx.x;  // B*3072 total
  const size_t n = t / 3072;
  const int    j = (int)(t % 3072);
  float v;
  if (j < 1024)      v = ctx[n * 1024 + j];
  else if (j < 2048) v = te [n * 1024 + j - 1024];
  else               v = ce [n * 1024 + j - 2048];
  E[t] = f2bf(v);
}

// z-perm + coupling split: cbuf[:, :128] = x_a (f32), xab = x_a (bf16), xb = x_b
__global__ __launch_bounds__(256)
void gather_kernel(const float* __restrict__ z,
                   const int* __restrict__ perm,
                   const int* __restrict__ idx_a,
                   const int* __restrict__ idx_b,
                   float* __restrict__ cbuf,
                   ushort_t* __restrict__ xab,
                   float* __restrict__ xb) {
  const int n = blockIdx.x;
  const int j = threadIdx.x;
  if (j < 128) {
    const int col = perm[idx_a[j]];
    const float v = z[(size_t)n * 256 + col];
    cbuf[(size_t)n * 256 + j] = v;
    xab [(size_t)n * 128 + j] = f2bf(v);
  } else {
    const int col = perm[idx_b[j - 128]];
    xb[(size_t)n * 128 + (j - 128)] = z[(size_t)n * 256 + col];
  }
}

// y_b = x_b * exp(scale) + shift ; logdet[n] += sum_j scale ; cbuf[:,128:]=y_b
__global__ __launch_bounds__(128)
void finish_kernel(const float* __restrict__ scale,
                   const float* __restrict__ shift,
                   const float* __restrict__ xb,
                   float* __restrict__ cbuf,
                   float* __restrict__ logdet) {
  const int n = blockIdx.x;
  const int j = threadIdx.x;
  float s = scale[(size_t)n * 128 + j];
  const float y = xb[(size_t)n * 128 + j] * __expf(s) + shift[(size_t)n * 128 + j];
  cbuf[(size_t)n * 256 + 128 + j] = y;
  __shared__ float red[4];
#pragma unroll
  for (int off = 16; off > 0; off >>= 1) s += __shfl_down(s, off, 32);
  if ((j & 31) == 0) red[j >> 5] = s;
  __syncthreads();
  if (j == 0) logdet[n] += red[0] + red[1] + red[2] + red[3];
}

// z_next[n, j] = concat[n, inv_pos[j]]
__global__ __launch_bounds__(256)
void scatter_kernel(const float* __restrict__ cbuf,
                    const int* __restrict__ inv,
                    float* __restrict__ zout) {
  const int t = blockIdx.x * 256 + threadIdx.x;
  const int n = t >> 8;
  const int j = t & 255;
  zout[(size_t)n * 256 + j] = cbuf[(size_t)n * 256 + inv[j]];
}

__global__ __launch_bounds__(256)
void zero_kernel(float* __restrict__ p, int n) {
  const int t = blockIdx.x * 256 + threadIdx.x;
  if (t < n) p[t] = 0.0f;
}

extern "C" void kernel_launch(void* const* d_in, const int* in_sizes, int n_in,
                              void* d_out, int out_size, void* d_ws, size_t ws_size,
                              hipStream_t stream) {
  (void)in_sizes; (void)n_in; (void)out_size; (void)ws_size;
  const float* x    = (const float*)d_in[0];
  const float* ctx  = (const float*)d_in[1];
  const float* te   = (const float*)d_in[2];
  const float* ce   = (const float*)d_in[3];
  const float* W1   = (const float*)d_in[4];
  const float* b1   = (const float*)d_in[5];
  const float* W2   = (const float*)d_in[6];
  const float* b2   = (const float*)d_in[7];
  const float* W3   = (const float*)d_in[8];
  const float* b3   = (const float*)d_in[9];
  const float* Ws   = (const float*)d_in[10];
  const float* bs   = (const float*)d_in[11];
  const float* Wt   = (const float*)d_in[12];
  const float* bt   = (const float*)d_in[13];
  const int*   perm = (const int*)d_in[14];
  const int*   ida  = (const int*)d_in[15];
  const int*   idb  = (const int*)d_in[16];
  const int*   inv  = (const int*)d_in[17];

  float* out    = (float*)d_out;
  float* logdet = out + (size_t)B_ * D_;

  // workspace carve-up (~208 MB)
  char* ws = (char*)d_ws;
  float*    xb    = (float*)ws;    ws += (size_t)B_ * 128 * sizeof(float);
  float*    cbuf  = (float*)ws;    ws += (size_t)B_ * 256 * sizeof(float);
  float*    scale = (float*)ws;    ws += (size_t)B_ * 128 * sizeof(float);
  float*    shift = (float*)ws;    ws += (size_t)B_ * 128 * sizeof(float);
  ushort_t* xab   = (ushort_t*)ws; ws += (size_t)B_ * 128 * sizeof(ushort_t);
  ushort_t* Eb    = (ushort_t*)ws; ws += (size_t)B_ * 3072 * sizeof(ushort_t);
  ushort_t* hA    = (ushort_t*)ws; ws += (size_t)B_ * H_ * sizeof(ushort_t);
  ushort_t* hB    = (ushort_t*)ws; ws += (size_t)B_ * H_ * sizeof(ushort_t);
  float*    zA    = (float*)ws;    ws += (size_t)B_ * D_ * sizeof(float);
  float*    zB    = (float*)ws;    ws += (size_t)B_ * D_ * sizeof(float);
  ushort_t* W1b   = (ushort_t*)ws; ws += (size_t)L_ * IN_ * H_ * sizeof(ushort_t);
  ushort_t* W2b   = (ushort_t*)ws; ws += (size_t)L_ * H_ * H_ * sizeof(ushort_t);
  ushort_t* W3b   = (ushort_t*)ws; ws += (size_t)L_ * H_ * H_ * sizeof(ushort_t);
  ushort_t* Wsb   = (ushort_t*)ws; ws += (size_t)L_ * H_ * BD_ * sizeof(ushort_t);
  ushort_t* Wtb   = (ushort_t*)ws;

  zero_kernel<<<(B_ + 255) / 256, 256, 0, stream>>>(logdet, B_);

  // one-time (per call) bf16 conversion / weight transposition
  wconv_kernel<<<dim3(H_ / 32, IN_ / 32, L_), 256, 0, stream>>>(W1, W1b, IN_, H_);
  wconv_kernel<<<dim3(H_ / 32, H_ / 32, L_), 256, 0, stream>>>(W2, W2b, H_, H_);
  wconv_kernel<<<dim3(H_ / 32, H_ / 32, L_), 256, 0, stream>>>(W3, W3b, H_, H_);
  wconv_kernel<<<dim3(BD_ / 32, H_ / 32, L_), 256, 0, stream>>>(Ws, Wsb, H_, BD_);
  wconv_kernel<<<dim3(BD_ / 32, H_ / 32, L_), 256, 0, stream>>>(Wt, Wtb, H_, BD_);
  econv_kernel<<<(B_ * 3072) / 256, 256, 0, stream>>>(ctx, te, ce, Eb);

  const float* zsrc = x;
  for (int l = 0; l < L_; ++l) {
    gather_kernel<<<B_, 256, 0, stream>>>(
        zsrc, perm + l * D_, ida + l * A_, idb + l * BD_, cbuf, xab, xb);

    // h1 = silu(concat @ W1 + b1)        (B x 3200) @ (3200 x 1024)
    fused_gemm_kernel<<<dim3(B_ / TILE_M, H_ / TILE_N), 256, 0, stream>>>(
        nullptr, xab, Eb,
        W1b + (size_t)l * IN_ * H_, b1 + (size_t)l * H_,
        hA, H_, IN_, /*act*/0, /*a_mode*/1);

    // h2 = silu(h1 @ W2 + b2)
    fused_gemm_kernel<<<dim3(B_ / TILE_M, H_ / TILE_N), 256, 0, stream>>>(
        hA, nullptr, nullptr,
        W2b + (size_t)l * H_ * H_, b2 + (size_t)l * H_,
        hB, H_, H_, 0, 0);

    // h3 = silu(h2 @ W3 + b3)
    fused_gemm_kernel<<<dim3(B_ / TILE_M, H_ / TILE_N), 256, 0, stream>>>(
        hB, nullptr, nullptr,
        W3b + (size_t)l * H_ * H_, b3 + (size_t)l * H_,
        hA, H_, H_, 0, 0);

    // scale = clip(h3 @ Ws + bs, +-3)
    fused_gemm_kernel<<<dim3(B_ / TILE_M, BD_ / TILE_N), 256, 0, stream>>>(
        hA, nullptr, nullptr,
        Wsb + (size_t)l * H_ * BD_, bs + (size_t)l * BD_,
        scale, BD_, H_, 1, 0);

    // shift = h3 @ Wt + bt
    fused_gemm_kernel<<<dim3(B_ / TILE_M, BD_ / TILE_N), 256, 0, stream>>>(
        hA, nullptr, nullptr,
        Wtb + (size_t)l * H_ * BD_, bt + (size_t)l * BD_,
        shift, BD_, H_, 2, 0);

    finish_kernel<<<B_, 128, 0, stream>>>(scale, shift, xb, cbuf, logdet);

    float* zdst = (l == L_ - 1) ? out : ((l & 1) ? zB : zA);
    scatter_kernel<<<(B_ * D_) / 256, 256, 0, stream>>>(cbuf, inv + l * D_, zdst);
    zsrc = zdst;
  }
}
